// SparseAttention_5179730559080
// MI455X (gfx1250) — matrix operand order repaired and checked by reading, not compiled, not run on hardware
//
#include <hip/hip_runtime.h>
#include <hip/hip_bf16.h>

typedef __bf16 bf16_t;
typedef __attribute__((ext_vector_type(16))) __bf16 v16bf;
typedef __attribute__((ext_vector_type(8)))  float  v8f;

#define S_LEN 4096
#define EMB   1024
#define NHEAD 16
#define HDIM  64
#define GSTEP 585   // global token positions: 585*j, j=0..7

union BF16x16 { v16bf v; uint32_t u[8]; };
union BF16U   { bf16_t h; unsigned short u; };

// ---------------------------------------------------------------------------
// CDNA5 async global->LDS DMA (ASYNCcnt path), inline asm per CDNA5 ISA.
// vdst = 32-bit LDS byte offset; vaddr = 64-bit global address (GV mode).
// ---------------------------------------------------------------------------
__device__ __forceinline__ void async_load_b128(uint32_t lds_off, const void* gaddr) {
  asm volatile("global_load_async_to_lds_b128 %0, %1, off"
               :: "v"(lds_off), "v"(gaddr)
               : "memory");
}
__device__ __forceinline__ void wait_async0() {
  asm volatile("s_wait_asynccnt 0" ::: "memory");
}
__device__ __forceinline__ uint32_t lds_off32(const void* p) {
  // LDS aperture: low 32 bits of the flat address are the LDS byte offset.
  return (uint32_t)(uintptr_t)p;
}

// ---------------------------------------------------------------------------
// WMMA helpers (wave32; A/B use contiguous K order, C/D layout per CDNA5 ISA 7.12.2)
// ---------------------------------------------------------------------------
__device__ __forceinline__ v8f wmma_bf16(v16bf a, v16bf b, v8f c) {
  return __builtin_amdgcn_wmma_f32_16x16x32_bf16(
      /*neg_a=*/false, a, /*neg_b=*/false, b,
      /*c_mod=*/(short)0, c, /*reuse_a=*/false, /*reuse_b=*/false);
}

// A-matrix 16x32 bf16 fragment from row-major LDS tile (pairs along K contiguous).
// lane L (<16): row=L, K = {0..15}; lane L(>=16): row=L-16, K = {16..31}
__device__ __forceinline__ v16bf load_fragA(const bf16_t* tile, int ldt,
                                            int row0, int k0, int lane) {
  BF16x16 f;
  const int r    = row0 + (lane & 15);
  const int kofs = (lane & 16) ? 16 : 0;
  const uint32_t* prow = (const uint32_t*)(tile + (size_t)r * ldt + k0);
#pragma unroll
  for (int j = 0; j < 8; ++j) {
    const int k = 2 * j + kofs;
    f.u[j] = prow[k >> 1];
  }
  return f.v;
}

// B-matrix 32x16 bf16 fragment from an LDS tile stored TRANSPOSED as [n][k]
// (so K-pairs are contiguous). lane L: col n = L%16, K = (L<16?0:16)+2j.
__device__ __forceinline__ v16bf load_fragB_T(const bf16_t* tileT, int ldt,
                                              int n0, int k0, int lane) {
  BF16x16 f;
  const int n    = n0 + (lane & 15);
  const int kofs = k0 + ((lane & 16) ? 16 : 0);
  const uint32_t* prow = (const uint32_t*)(tileT + (size_t)n * ldt + kofs);
#pragma unroll
  for (int j = 0; j < 8; ++j) f.u[j] = prow[j];
  return f.v;
}

// B-matrix 32x16 bf16 fragment from a ROW-MAJOR [k][n] LDS tile (strided k).
__device__ __forceinline__ v16bf load_fragB_RM(const bf16_t* tile, int ldt,
                                               int k0, int n0, int lane) {
  BF16x16 f;
  const int n    = n0 + (lane & 15);
  const int kofs = k0 + ((lane & 16) ? 16 : 0);
#pragma unroll
  for (int j = 0; j < 8; ++j) {
    BF16U lo, hi;
    lo.h = tile[(size_t)(kofs + 2 * j) * ldt + n];
    hi.h = tile[(size_t)(kofs + 2 * j + 1) * ldt + n];
    f.u[j] = (uint32_t)lo.u | ((uint32_t)hi.u << 16);
  }
  return f.v;
}

// ---------------------------------------------------------------------------
// Elementwise converts
// ---------------------------------------------------------------------------
__global__ void k_cvt_bf16(const float* __restrict__ s, bf16_t* __restrict__ d, int n) {
  for (int i = blockIdx.x * blockDim.x + threadIdx.x; i < n;
       i += gridDim.x * blockDim.x)
    d[i] = (bf16_t)s[i];
}

__global__ void k_pack_wqkv(const float* __restrict__ Wq, const float* __restrict__ Wk,
                            const float* __restrict__ Wv, bf16_t* __restrict__ Wqkv) {
  const int n = EMB * 3 * EMB;
  for (int i = blockIdx.x * blockDim.x + threadIdx.x; i < n;
       i += gridDim.x * blockDim.x) {
    const int c = i % (3 * EMB);
    const int k = i / (3 * EMB);
    const int which = c >> 10;
    const int cc = c & (EMB - 1);
    const float* W = (which == 0) ? Wq : (which == 1) ? Wk : Wv;
    Wqkv[i] = (bf16_t)W[(size_t)k * EMB + cc];
  }
}

// ---------------------------------------------------------------------------
// Tiled bf16 WMMA GEMM: C[M,N] = A[M,K] * B[K,N]. 128x128x32 block tile,
// 256 threads = 8 waves (2x4), each wave owns a 64x32 patch (8 WMMA tiles).
// MODE 0: QKV epilogue (bias add, Q*=1/8, scatter to head-major bf16 Q/K/V)
// MODE 1: fp32 output + bias
// ---------------------------------------------------------------------------
template <int MODE>
__global__ __launch_bounds__(256) void gemm_bf16_wmma(
    const bf16_t* __restrict__ A, const bf16_t* __restrict__ B,
    int M, int N, int K,
    const float* __restrict__ b0, const float* __restrict__ b1,
    const float* __restrict__ b2,
    bf16_t* __restrict__ Qb, bf16_t* __restrict__ Kb, bf16_t* __restrict__ Vb,
    float* __restrict__ OutF) {
  __shared__ __align__(16) bf16_t As[128][40];   // [m][k], padded
  __shared__ __align__(16) bf16_t BsT[128][40];  // [n][k], transposed, padded

  const int tid  = threadIdx.x;
  const int lane = tid & 31;
  const int wave = tid >> 5;
  const int wm = wave >> 2;  // 0..1
  const int wn = wave & 3;   // 0..3
  const int rowBase = blockIdx.y * 128;
  const int colBase = blockIdx.x * 128;

  v8f acc[4][2];
  const v8f vzero = {};
#pragma unroll
  for (int i = 0; i < 4; ++i)
#pragma unroll
    for (int j = 0; j < 2; ++j) acc[i][j] = vzero;

  for (int k0 = 0; k0 < K; k0 += 32) {
    {  // stage A tile (row-major)
      const int r = tid >> 2, kk = (tid & 3) * 8;
#pragma unroll
      for (int p = 0; p < 2; ++p) {
        const uint4 v = *(const uint4*)(A + (size_t)(rowBase + r + p * 64) * K + k0 + kk);
        *(uint4*)(&As[r + p * 64][kk]) = v;
      }
    }
    {  // stage B tile transposed -> [n][k]
      const int kr = tid >> 4, cc = (tid & 15) * 8;
#pragma unroll
      for (int p = 0; p < 2; ++p) {
        const int krow = kr + p * 16;
        const bf16_t* src = B + (size_t)(k0 + krow) * N + colBase + cc;
#pragma unroll
        for (int i = 0; i < 8; ++i) BsT[cc + i][krow] = src[i];
      }
    }
    // prefetch next k-step tiles into cache while this step computes
    if (k0 + 32 < K) {
      __builtin_prefetch(A + (size_t)(rowBase + (tid >> 2)) * K + k0 + 32 + (tid & 3) * 8, 0, 1);
      __builtin_prefetch(B + (size_t)(k0 + 32 + (tid >> 4)) * N + colBase + (tid & 15) * 8, 0, 1);
    }
    __syncthreads();

    v16bf bf[2];
#pragma unroll
    for (int j = 0; j < 2; ++j)
      bf[j] = load_fragB_T(&BsT[0][0], 40, wn * 32 + j * 16, 0, lane);
#pragma unroll
    for (int i = 0; i < 4; ++i) {
      const v16bf af = load_fragA(&As[0][0], 40, wm * 64 + i * 16, 0, lane);
#pragma unroll
      for (int j = 0; j < 2; ++j) acc[i][j] = wmma_bf16(af, bf[j], acc[i][j]);
    }
    __syncthreads();
  }

  // Epilogue: C/D layout — VGPR r: M = r + (lane>=16 ? 8 : 0), N = lane%16
#pragma unroll
  for (int i = 0; i < 4; ++i)
#pragma unroll
    for (int j = 0; j < 2; ++j)
#pragma unroll
      for (int r = 0; r < 8; ++r) {
        const int m = rowBase + wm * 64 + i * 16 + r + ((lane & 16) ? 8 : 0);
        const int n = colBase + wn * 32 + j * 16 + (lane & 15);
        const float v = acc[i][j][r];
        if (MODE == 0) {
          const int which = n >> 10;
          const int cc = n & (EMB - 1);
          const float bias = ((which == 0) ? b0 : (which == 1) ? b1 : b2)[cc];
          float val = v + bias;
          if (which == 0) val *= 0.125f;  // fold 1/sqrt(64) into Q
          const int head = cc >> 6, d = cc & 63;
          bf16_t* dst = (which == 0) ? Qb : (which == 1) ? Kb : Vb;
          dst[((size_t)head * S_LEN + m) * HDIM + d] = (bf16_t)val;
        } else {
          OutF[(size_t)m * N + n] = v + b0[n];
        }
      }
}

// ---------------------------------------------------------------------------
// Local sparse attention: per (query block of 64, head). Columns = 64 local
// + 8 global (padded to 16; global cols inside the local block are masked to
// avoid double counting). Scores & PV via WMMA; softmax in LDS.
// Bulk tile staging uses CDNA5 async global->LDS DMA.
// ---------------------------------------------------------------------------
__global__ __launch_bounds__(256) void attn_local(
    const bf16_t* __restrict__ Qb, const bf16_t* __restrict__ Kb,
    const bf16_t* __restrict__ Vb, bf16_t* __restrict__ AO) {
  const int qb = blockIdx.x;  // 0..63
  const int h  = blockIdx.y;  // 0..15
  const int tid = threadIdx.x, lane = tid & 31, wave = tid >> 5;

  // R0 holds Qs(9216B)+Ks(11520B) during score phase, reused as P(13312B).
  __shared__ __align__(16) char   R0[9216 + 11520];
  __shared__ __align__(16) bf16_t Vs[96][72];
  __shared__ float Sf[64][81];

  bf16_t(*Qs)[72] = (bf16_t(*)[72])R0;
  bf16_t(*Ks)[72] = (bf16_t(*)[72])(R0 + 9216);

  const int s0 = qb * 64;
  const bf16_t* Qh = Qb + (size_t)h * S_LEN * HDIM;
  const bf16_t* Kh = Kb + (size_t)h * S_LEN * HDIM;
  const bf16_t* Vh = Vb + (size_t)h * S_LEN * HDIM;

  {  // stage Q / K_local / V_local (64x64 each): async DMA, 16B per lane-op
    const int r = tid >> 3, cc = (tid & 7) * 8;
#pragma unroll
    for (int p = 0; p < 2; ++p) {
      const int rr = r + p * 32;
      async_load_b128(lds_off32(&Qs[rr][cc]), Qh + (size_t)(s0 + rr) * HDIM + cc);
      async_load_b128(lds_off32(&Ks[rr][cc]), Kh + (size_t)(s0 + rr) * HDIM + cc);
      async_load_b128(lds_off32(&Vs[rr][cc]), Vh + (size_t)(s0 + rr) * HDIM + cc);
    }
  }
  {  // stage 16 global rows (8 valid) + zero pad rows of Vs (regular path)
    const uint4 z = {0u, 0u, 0u, 0u};
    if (tid < 128) {
      const int j = tid >> 3, cc = (tid & 7) * 8;
      uint4 kv = z, vv = z;
      if (j < 8) {
        const int g = GSTEP * j;
        kv = *(const uint4*)(Kh + (size_t)g * HDIM + cc);
        vv = *(const uint4*)(Vh + (size_t)g * HDIM + cc);
      }
      *(uint4*)(&Ks[64 + j][cc]) = kv;
      *(uint4*)(&Vs[64 + j][cc]) = vv;
    } else {
      const int j = (tid - 128) >> 3, cc = ((tid - 128) & 7) * 8;
      *(uint4*)(&Vs[80 + j][cc]) = z;
    }
  }
  wait_async0();   // drain this wave's ASYNCcnt before the barrier
  __syncthreads();

  // Phase B: scores S[64][80] = Q * K^T   (Q already carries 1/8 scale)
  for (int t = wave; t < 20; t += 8) {
    const int tm = t / 5, tn = t % 5;
    v8f acc = {};
#pragma unroll
    for (int kk = 0; kk < 64; kk += 32) {
      const v16bf a = load_fragA(&Qs[0][0], 72, tm * 16, kk, lane);
      const v16bf b = load_fragB_T(&Ks[0][0], 72, tn * 16, kk, lane);
      acc = wmma_bf16(a, b, acc);
    }
#pragma unroll
    for (int r = 0; r < 8; ++r) {
      const int m = tm * 16 + r + ((lane & 16) ? 8 : 0);
      const int n = tn * 16 + (lane & 15);
      float v = acc[r];
      if (n >= 64) {  // global slab: only j<8 valid, dedupe cols inside block
        const int j = n - 64;
        const bool valid = (j < 8) && (((GSTEP * j) >> 6) != qb);
        if (!valid) v = -3.0e38f;
      }
      Sf[m][n] = v;
    }
  }
  __syncthreads();

  // Phase C: row softmax over 80 cols -> P bf16 [64][96] (last 16 cols zero)
  bf16_t(*P)[104] = (bf16_t(*)[104])R0;
  if (tid < 64) {
    float mx = -3.4e38f;
#pragma unroll 4
    for (int c = 0; c < 80; ++c) mx = fmaxf(mx, Sf[tid][c]);
    float sum = 0.f;
#pragma unroll 4
    for (int c = 0; c < 80; ++c) {
      const float e = __expf(Sf[tid][c] - mx);
      Sf[tid][c] = e;
      sum += e;
    }
    const float inv = 1.f / sum;
#pragma unroll 4
    for (int c = 0; c < 80; ++c) P[tid][c] = (bf16_t)(Sf[tid][c] * inv);
#pragma unroll
    for (int c = 80; c < 96; ++c) P[tid][c] = (bf16_t)0.0f;
  }
  __syncthreads();

  // Phase D: O[64][64] = P[64][96] @ Vs[96][64]
  for (int t = wave; t < 16; t += 8) {
    const int tm = t >> 2, tn = t & 3;
    v8f acc = {};
#pragma unroll
    for (int kk = 0; kk < 96; kk += 32) {
      const v16bf a = load_fragA(&P[0][0], 104, tm * 16, kk, lane);
      const v16bf b = load_fragB_RM(&Vs[0][0], 72, kk, tn * 16, lane);
      acc = wmma_bf16(a, b, acc);
    }
#pragma unroll
    for (int r = 0; r < 8; ++r) {
      const int m = tm * 16 + r + ((lane & 16) ? 8 : 0);
      const int d = tn * 16 + (lane & 15);
      AO[(size_t)(s0 + m) * EMB + h * HDIM + d] = (bf16_t)acc[r];
    }
  }
}

// ---------------------------------------------------------------------------
// Global query rows: the 8 global tokens attend to ALL 4096 keys.
// One block per (global idx j, head). Overwrites those rows of AO.
// ---------------------------------------------------------------------------
__global__ __launch_bounds__(256) void attn_global(
    const bf16_t* __restrict__ Qb, const bf16_t* __restrict__ Kb,
    const bf16_t* __restrict__ Vb, bf16_t* __restrict__ AO) {
  const int j = blockIdx.x;  // 0..7
  const int h = blockIdx.y;  // 0..15
  const int p = GSTEP * j;
  const int tid = threadIdx.x;

  __shared__ float sc[S_LEN];
  __shared__ float qrow[HDIM];
  __shared__ float red[256];

  const bf16_t* Qh = Qb + (size_t)h * S_LEN * HDIM;
  const bf16_t* Kh = Kb + (size_t)h * S_LEN * HDIM;
  const bf16_t* Vh = Vb + (size_t)h * S_LEN * HDIM;

  if (tid < HDIM) qrow[tid] = (float)Qh[(size_t)p * HDIM + tid];
  __syncthreads();

  float lmax = -3.4e38f;
  for (int c = tid; c < S_LEN; c += 256) {
    float dot = 0.f;
#pragma unroll 8
    for (int d = 0; d < HDIM; ++d) dot += qrow[d] * (float)Kh[(size_t)c * HDIM + d];
    sc[c] = dot;  // Q already scaled by 1/8
    lmax = fmaxf(lmax, dot);
  }
  red[tid] = lmax;
  __syncthreads();
  for (int s = 128; s > 0; s >>= 1) {
    if (tid < s) red[tid] = fmaxf(red[tid], red[tid + s]);
    __syncthreads();
  }
  const float mx = red[0];
  __syncthreads();

  float lsum = 0.f;
  for (int c = tid; c < S_LEN; c += 256) {
    const float e = __expf(sc[c] - mx);
    sc[c] = e;
    lsum += e;
  }
  red[tid] = lsum;
  __syncthreads();
  for (int s = 128; s > 0; s >>= 1) {
    if (tid < s) red[tid] += red[tid + s];
    __syncthreads();
  }
  const float inv = 1.f / red[0];
  __syncthreads();

  // out[d] = inv * sum_c sc[c]*V[c][d]; 4 partial chunks of 1024 cols
  const int d = tid & 63;
  const int part = tid >> 6;
  float acc = 0.f;
  const int c0 = part * 1024;
  for (int c = c0; c < c0 + 1024; ++c)
    acc += sc[c] * (float)Vh[(size_t)c * HDIM + d];
  red[tid] = acc;
  __syncthreads();
  if (tid < 64) {
    const float o = (red[tid] + red[tid + 64] + red[tid + 128] + red[tid + 192]) * inv;
    AO[(size_t)p * EMB + h * HDIM + tid] = (bf16_t)o;
  }
}

// ---------------------------------------------------------------------------
// Host launcher
// ---------------------------------------------------------------------------
extern "C" void kernel_launch(void* const* d_in, const int* in_sizes, int n_in,
                              void* d_out, int out_size, void* d_ws, size_t ws_size,
                              hipStream_t stream) {
  const float* x  = (const float*)d_in[0];
  const float* Wq = (const float*)d_in[1];
  const float* bq = (const float*)d_in[2];
  const float* Wk = (const float*)d_in[3];
  const float* bk = (const float*)d_in[4];
  const float* Wv = (const float*)d_in[5];
  const float* bv = (const float*)d_in[6];
  const float* Wo = (const float*)d_in[7];
  const float* bo = (const float*)d_in[8];
  float* out = (float*)d_out;

  char* ws = (char*)d_ws;
  bf16_t* xb   = (bf16_t*)(ws);                         // 8 MB
  bf16_t* wqkv = (bf16_t*)(ws + 8388608);               // 6 MB
  bf16_t* wob  = (bf16_t*)(ws + 14680064);              // 2 MB
  bf16_t* Qb   = (bf16_t*)(ws + 16777216);              // 8 MB
  bf16_t* Kb   = (bf16_t*)(ws + 25165824);              // 8 MB
  bf16_t* Vb   = (bf16_t*)(ws + 33554432);              // 8 MB
  bf16_t* AO   = (bf16_t*)(ws + 41943040);              // 8 MB  (total 48 MB)

  // 1) precision converts
  k_cvt_bf16<<<1024, 256, 0, stream>>>(x, xb, S_LEN * EMB);
  k_pack_wqkv<<<1024, 256, 0, stream>>>(Wq, Wk, Wv, wqkv);
  k_cvt_bf16<<<512, 256, 0, stream>>>(Wo, wob, EMB * EMB);

  // 2) fused QKV projection: [4096,3072] = xb @ wqkv (+bias, Q scaled 1/8)
  gemm_bf16_wmma<0><<<dim3(3 * EMB / 128, S_LEN / 128), 256, 0, stream>>>(
      xb, wqkv, S_LEN, 3 * EMB, EMB, bq, bk, bv, Qb, Kb, Vb, nullptr);

  // 3) local + global-column sparse attention
  attn_local<<<dim3(S_LEN / 64, NHEAD), 256, 0, stream>>>(Qb, Kb, Vb, AO);

  // 4) full attention for the 8 global query rows (overwrite)
  attn_global<<<dim3(8, NHEAD), 256, 0, stream>>>(Qb, Kb, Vb, AO);

  // 5) output projection: out = AO @ Wo + bo (fp32)
  gemm_bf16_wmma<1><<<dim3(EMB / 128, S_LEN / 128), 256, 0, stream>>>(
      AO, wob, S_LEN, EMB, EMB, bo, nullptr, nullptr,
      nullptr, nullptr, nullptr, out);
}
